// CMS_Block_63917703299770
// MI455X (gfx1250) — matrix operand order repaired and checked by reading, not compiled, not run on hardware
//
#include <hip/hip_runtime.h>
#include <hip/hip_bf16.h>
#include <math.h>

typedef __bf16 bf16;
typedef __attribute__((ext_vector_type(16))) __bf16 bf16x16;
typedef __attribute__((ext_vector_type(8)))  __bf16 bf16x8;
typedef __attribute__((ext_vector_type(8)))  float  f32x8;

#define MTOT 65536   // B * H * W = 16 * 4096
#define NPB  4096    // H * W per batch

__device__ __forceinline__ bf16x16 combineA(bf16x8 lo, bf16x8 hi) {
    return __builtin_shufflevector(lo, hi, 0,1,2,3,4,5,6,7,8,9,10,11,12,13,14,15);
}
__device__ __forceinline__ f32x8 wmma_bf16(bf16x16 a, bf16x16 b, f32x8 c) {
    return __builtin_amdgcn_wmma_f32_16x16x32_bf16(false, a, false, b, (short)0, c, false, false);
}

// ---------------------------------------------------------------------------
// Generic GEMM: out[orow+o, m] = sum_k W[o,k] * B[m, coff+k]
// W: fp32 row-major [O,K] (K<=256). B: bf16 row-major [MTOT, ldb].
// grid = (MTOT/512, O/32); block = 256 (8 waves).
// Each wave: 32(o) x 64(m) tile -> every B fragment feeds 2 WMMAs.
// ---------------------------------------------------------------------------
__global__ __launch_bounds__(256)
void gemm_wmma(const float* __restrict__ W, int K,
               const bf16* __restrict__ Bm, int ldb, int coff,
               float* __restrict__ out, int orow)
{
    __shared__ bf16 ldsW[32 * 256];
    const int tid = threadIdx.x;
    const int o0  = blockIdx.y * 32;
    for (int i = tid; i < 32 * K; i += 256)
        ldsW[i] = (bf16)W[(size_t)(o0 + i / K) * K + (i % K)];
    __syncthreads();

    const int wave = tid >> 5, lane = tid & 31;
    const int row  = lane & 15, half = lane >> 4;
    const size_t m0 = ((size_t)blockIdx.x * 8 + wave) * 64;

    f32x8 acc[2][4] = {};
    for (int k0 = 0; k0 < K; k0 += 32) {
        bf16x16 a[2];
        #pragma unroll
        for (int u = 0; u < 2; ++u) {
            const bf16* wp = &ldsW[(u * 16 + row) * K + k0 + 16 * half];
            a[u] = combineA(*(const bf16x8*)wp, *(const bf16x8*)(wp + 8));
        }
        const bf16* bp = Bm + (size_t)(m0 + row) * ldb + coff + k0 + 16 * half;
        #pragma unroll
        for (int t = 0; t < 4; ++t) {
            bf16x16 b = *(const bf16x16*)(bp + (size_t)t * 16 * ldb);
            acc[0][t] = wmma_bf16(a[0], b, acc[0][t]);
            acc[1][t] = wmma_bf16(a[1], b, acc[1][t]);
        }
    }
    #pragma unroll
    for (int u = 0; u < 2; ++u)
        #pragma unroll
        for (int t = 0; t < 4; ++t) {
            size_t m = m0 + t * 16 + row;
            #pragma unroll
            for (int j = 0; j < 8; ++j)
                out[(size_t)(orow + o0 + u * 16 + j + 8 * half) * MTOT + m] = acc[u][t][j];
        }
}

// ---------------------------------------------------------------------------
// 3x3 SAME conv as implicit GEMM over 9 taps, 64->64 channels.
// W3: [64,64,3,3] fp32. ht: [MTOT,256] bf16, channel slice at coff.
// grid = (512, 2); each wave: 32(o) x 16(m) tile (m tile = one row segment),
// so every B fragment feeds 2 WMMAs.
// ---------------------------------------------------------------------------
__global__ __launch_bounds__(256)
void conv3x3_wmma(const float* __restrict__ W3, const bf16* __restrict__ ht,
                  float* __restrict__ out, int orow, int coff)
{
    __shared__ bf16 ldsA[32 * 9 * 64];   // [o][tap][c], 36 KB
    const int tid = threadIdx.x;
    const int o0  = blockIdx.y * 32;
    for (int i = tid; i < 32 * 9 * 64; i += 256) {
        int o = i / (9 * 64), rem = i % (9 * 64);
        int tap = rem / 64, c = rem % 64;
        ldsA[i] = (bf16)W3[((size_t)(o0 + o) * 64 + c) * 9 + tap];
    }
    __syncthreads();

    const int wave = tid >> 5, lane = tid & 31;
    const int row  = lane & 15, half = lane >> 4;
    const int mtile = blockIdx.x * 8 + wave;
    const int m0 = mtile * 16;
    const int batch = m0 >> 12;
    const int pos = m0 & 4095;
    const int y = pos >> 6, xcol = (pos & 63) + row;

    f32x8 acc[2] = {};
    for (int tap = 0; tap < 9; ++tap) {
        int dy = tap / 3 - 1, dx = tap % 3 - 1;
        int yy = y + dy;
        if (yy < 0 || yy >= 64) continue;           // wave-uniform
        int xx = xcol + dx;
        bool valid = (xx >= 0) && (xx < 64);        // per-lane (column) validity
        size_t mm = (size_t)batch * NPB + (size_t)yy * 64 + xx;
        #pragma unroll
        for (int k0 = 0; k0 < 64; k0 += 32) {
            bf16x16 b = {};
            if (valid) b = *(const bf16x16*)(ht + mm * 256 + coff + k0 + 16 * half);
            #pragma unroll
            for (int u = 0; u < 2; ++u) {
                const bf16* wp = &ldsA[((u * 16 + row) * 9 + tap) * 64 + k0 + 16 * half];
                bf16x16 a = combineA(*(const bf16x8*)wp, *(const bf16x8*)(wp + 8));
                acc[u] = wmma_bf16(a, b, acc[u]);   // EXEC full here
            }
        }
    }
    #pragma unroll
    for (int u = 0; u < 2; ++u)
        #pragma unroll
        for (int j = 0; j < 8; ++j)
            out[(size_t)(orow + o0 + u * 16 + j + 8 * half) * MTOT + m0 + row] = acc[u][j];
}

// ---------------------------------------------------------------------------
// ctx[c,d] = sum_n k[c,n] * v[d,n] per batch; writes ctxT[b][d][c] bf16.
// kbf/vbf: [64, MTOT] bf16. grid = 32 (=16 batches * 16 tiles / 8 waves).
// ---------------------------------------------------------------------------
__global__ __launch_bounds__(256)
void ctx_gemm(const bf16* __restrict__ kbf, const bf16* __restrict__ vbf,
              bf16* __restrict__ ctxT)
{
    const int tid = threadIdx.x, wave = tid >> 5, lane = tid & 31;
    const int wg = blockIdx.x * 8 + wave;
    const int batch = wg >> 4, tile = wg & 15;
    const int ct = tile >> 2, dt = tile & 3;
    const int row = lane & 15, half = lane >> 4;
    const size_t base = (size_t)batch * NPB;

    f32x8 acc = {};
    for (int n0 = 0; n0 < NPB; n0 += 32) {
        const bf16* ap = kbf + (size_t)(ct * 16 + row) * MTOT + base + n0 + 16 * half;
        bf16x16 a = combineA(*(const bf16x8*)ap, *(const bf16x8*)(ap + 8));
        bf16x16 b = *(const bf16x16*)(vbf + (size_t)(dt * 16 + row) * MTOT + base + n0 + 16 * half);
        acc = wmma_bf16(a, b, acc);
    }
    #pragma unroll
    for (int j = 0; j < 8; ++j) {
        int c = ct * 16 + j + 8 * half;
        int d = dt * 16 + row;
        ctxT[(size_t)batch * 4096 + d * 64 + c] = (bf16)acc[j];
    }
}

// ---------------------------------------------------------------------------
// att[n,d] = sum_c qT[n,c] * ctxT[d,c] per batch; writes att row-major [m,64].
// grid = 512 (=16 batches * 256 n-tiles / 8 waves).
// ---------------------------------------------------------------------------
__global__ __launch_bounds__(256)
void att_gemm(const bf16* __restrict__ qT, const bf16* __restrict__ ctxT,
              bf16* __restrict__ att)
{
    const int tid = threadIdx.x, wave = tid >> 5, lane = tid & 31;
    const int wg = blockIdx.x * 8 + wave;
    const int batch = wg >> 8;
    const int n0 = (wg & 255) * 16;
    const int row = lane & 15, half = lane >> 4;

    f32x8 acc[4] = {};
    #pragma unroll
    for (int k0 = 0; k0 < 64; k0 += 32) {
        const bf16* ap = qT + ((size_t)batch * NPB + n0 + row) * 64 + k0 + 16 * half;
        bf16x16 a = combineA(*(const bf16x8*)ap, *(const bf16x8*)(ap + 8));
        #pragma unroll
        for (int dt = 0; dt < 4; ++dt) {
            bf16x16 b = *(const bf16x16*)(ctxT + (size_t)batch * 4096 + (dt * 16 + row) * 64 + k0 + 16 * half);
            acc[dt] = wmma_bf16(a, b, acc[dt]);
        }
    }
    #pragma unroll
    for (int dt = 0; dt < 4; ++dt)
        #pragma unroll
        for (int j = 0; j < 8; ++j)
            att[((size_t)batch * NPB + n0 + j + 8 * half) * 64 + dt * 16 + row] = (bf16)acc[dt][j];
}

// ---------------------------------------------------------------------------
// LDS-tiled transpose fp32 [ch, m] -> bf16 [m, nch], optional BN(+ReLU) fused.
// batched=1 reads NCHW layout (x input). block = (32, 8).
// ---------------------------------------------------------------------------
__global__ __launch_bounds__(256)
void transpose_cvt(const float* __restrict__ src, int batched, int Cfull, int chBase,
                   int nch, bf16* __restrict__ dst,
                   const float* __restrict__ mean, const float* __restrict__ rs,
                   const float* __restrict__ g, const float* __restrict__ b, int relu)
{
    __shared__ float tile[32][33];
    const int m0 = blockIdx.x * 32;
    const int ch0 = blockIdx.y * 32;
    const int tx = threadIdx.x, ty = threadIdx.y;
    #pragma unroll
    for (int r = 0; r < 4; ++r) {
        int ch = ch0 + ty + r * 8;
        int m = m0 + tx;
        size_t idx = batched
            ? ((size_t)(m >> 12) * Cfull + chBase + ch) * NPB + (m & 4095)
            : (size_t)(chBase + ch) * MTOT + m;
        float v = src[idx];
        if (mean) v = (v - mean[ch]) * rs[ch] * g[ch] + b[ch];
        if (relu) v = fmaxf(v, 0.f);
        tile[ty + r * 8][tx] = v;
    }
    __syncthreads();
    #pragma unroll
    for (int r = 0; r < 4; ++r) {
        int m = m0 + ty + r * 8;
        int ch = ch0 + tx;
        dst[(size_t)m * nch + ch] = (bf16)tile[tx][ty + r * 8];
    }
}

// BN statistics per channel: mean and rsqrt(var+eps). One block per channel.
__global__ __launch_bounds__(256)
void bn_stats(const float* __restrict__ buf, float* __restrict__ mean, float* __restrict__ rs)
{
    __shared__ float s1[256], s2[256];
    const int ch = blockIdx.x, tid = threadIdx.x;
    const float* p = buf + (size_t)ch * MTOT;
    float a = 0.f, b = 0.f;
    for (int i = tid; i < MTOT; i += 256) { float v = p[i]; a += v; b += v * v; }
    s1[tid] = a; s2[tid] = b; __syncthreads();
    for (int s = 128; s > 0; s >>= 1) {
        if (tid < s) { s1[tid] += s1[tid + s]; s2[tid] += s2[tid + s]; }
        __syncthreads();
    }
    if (tid == 0) {
        float m = s1[0] * (1.f / MTOT);
        float v = s2[0] * (1.f / MTOT) - m * m;
        mean[ch] = m;
        rs[ch] = rsqrtf(v + 1e-5f);
    }
}

// In-place BN on fp32 [256, MTOT] (residual path, no ReLU).
__global__ __launch_bounds__(256)
void bn_inplace(float* __restrict__ buf, const float* __restrict__ mean,
                const float* __restrict__ rs, const float* __restrict__ g,
                const float* __restrict__ b)
{
    const size_t total = (size_t)256 * MTOT;
    for (size_t i = (size_t)blockIdx.x * blockDim.x + threadIdx.x; i < total;
         i += (size_t)gridDim.x * blockDim.x) {
        int ch = (int)(i >> 16);
        buf[i] = (buf[i] - mean[ch]) * rs[ch] * g[ch] + b[ch];
    }
}

// Softmax over N=4096 for each (batch, k-channel); reads qkv rows 64..127 fp32,
// writes bf16 [64, MTOT]. One block per (batch, channel).
__global__ __launch_bounds__(256)
void softmax_rows(const float* __restrict__ qkv, bf16* __restrict__ kbf)
{
    __shared__ float red[256];
    const int kc = blockIdx.x & 63, batch = blockIdx.x >> 6;
    const int tid = threadIdx.x;
    const float* row = qkv + (size_t)(64 + kc) * MTOT + (size_t)batch * NPB;
    bf16* orow = kbf + (size_t)kc * MTOT + (size_t)batch * NPB;

    float mx = -INFINITY;
    for (int i = tid; i < NPB; i += 256) mx = fmaxf(mx, row[i]);
    red[tid] = mx; __syncthreads();
    for (int s = 128; s > 0; s >>= 1) {
        if (tid < s) red[tid] = fmaxf(red[tid], red[tid + s]);
        __syncthreads();
    }
    mx = red[0]; __syncthreads();

    float sum = 0.f;
    for (int i = tid; i < NPB; i += 256) sum += expf(row[i] - mx);
    red[tid] = sum; __syncthreads();
    for (int s = 128; s > 0; s >>= 1) {
        if (tid < s) red[tid] += red[tid + s];
        __syncthreads();
    }
    float inv = 1.f / red[0];
    for (int i = tid; i < NPB; i += 256)
        orow[i] = (bf16)(expf(row[i] - mx) * inv);
}

__global__ __launch_bounds__(256)
void cvt_f32_bf16(const float* __restrict__ src, bf16* __restrict__ dst, size_t n)
{
    for (size_t i = (size_t)blockIdx.x * blockDim.x + threadIdx.x; i < n;
         i += (size_t)gridDim.x * blockDim.x)
        dst[i] = (bf16)src[i];
}

// Gather 4x64 per-channel params into one 256 array.
__global__ void concat4(const float* __restrict__ a, const float* __restrict__ b,
                        const float* __restrict__ c, const float* __restrict__ d,
                        float* __restrict__ o)
{
    int t = threadIdx.x;
    float v = (t < 64) ? a[t] : (t < 128) ? b[t - 64] : (t < 192) ? c[t - 128] : d[t - 192];
    o[t] = v;
}

// out = relu(bn(F) + R), written in NCHW order.
__global__ __launch_bounds__(256)
void final_fuse(const float* __restrict__ F, const float* __restrict__ R,
                const float* __restrict__ mean, const float* __restrict__ rs,
                const float* __restrict__ g, const float* __restrict__ b,
                float* __restrict__ out)
{
    const size_t total = (size_t)256 * MTOT;
    for (size_t i = (size_t)blockIdx.x * blockDim.x + threadIdx.x; i < total;
         i += (size_t)gridDim.x * blockDim.x) {
        int ch = (int)(i >> 16);
        int m = (int)(i & 65535);
        int bt = m >> 12, n = m & 4095;
        float v = (F[i] - mean[ch]) * rs[ch] * g[ch] + b[ch] + R[i];
        out[((size_t)bt * 256 + ch) * NPB + n] = fmaxf(v, 0.f);
    }
}

// ---------------------------------------------------------------------------
extern "C" void kernel_launch(void* const* d_in, const int* in_sizes, int n_in,
                              void* d_out, int out_size, void* d_ws, size_t ws_size,
                              hipStream_t stream)
{
    const float* x     = (const float*)d_in[0];
    const float* W_sc  = (const float*)d_in[1];
    const float* g_sc  = (const float*)d_in[2];
    const float* b_sc  = (const float*)d_in[3];
    const float* W1    = (const float*)d_in[4];
    const float* g1    = (const float*)d_in[5];
    const float* b1    = (const float*)d_in[6];
    const float* Wb0   = (const float*)d_in[7];
    const float* gb0   = (const float*)d_in[8];
    const float* bb0   = (const float*)d_in[9];
    const float* Wb1   = (const float*)d_in[10];
    const float* gb1   = (const float*)d_in[11];
    const float* bb1   = (const float*)d_in[12];
    const float* Wqkv2 = (const float*)d_in[13];
    const float* Wproj2= (const float*)d_in[14];
    const float* ga2   = (const float*)d_in[15];
    const float* ba2   = (const float*)d_in[16];
    const float* Wqkv3 = (const float*)d_in[17];
    const float* Wproj3= (const float*)d_in[18];
    const float* ga3   = (const float*)d_in[19];
    const float* ba3   = (const float*)d_in[20];
    const float* W2    = (const float*)d_in[21];
    const float* g2    = (const float*)d_in[22];
    const float* b2    = (const float*)d_in[23];

    const size_t MB = 1024 * 1024;
    char* ws = (char*)d_ws;
    bf16*  xt   = (bf16*) (ws + 0);            // [MTOT,128] bf16     16 MB
    bf16*  ht   = (bf16*) (ws + 16 * MB);      // [MTOT,256] bf16     32 MB
    float* bufA = (float*)(ws + 48 * MB);      // [256,MTOT] f32      64 MB (Y1 / branches / F)
    float* bufR = (float*)(ws + 112 * MB);     // [256,MTOT] f32      64 MB (residual)
    float* qkv  = (float*)(ws + 176 * MB);     // [192,MTOT] f32      48 MB
    bf16*  ot   = (bf16*) (ws + 176 * MB);     // [MTOT,256] bf16 (reuses qkv region)
    bf16*  kbf  = (bf16*) (ws + 224 * MB);     // [64,MTOT] bf16       8 MB
    bf16*  vbf  = (bf16*) (ws + 232 * MB);     // [64,MTOT] bf16       8 MB
    bf16*  qT   = (bf16*) (ws + 240 * MB);     // [MTOT,64] bf16       8 MB
    bf16*  attb = (bf16*) (ws + 248 * MB);     // [MTOT,64] bf16       8 MB
    bf16*  ctxT = (bf16*) (ws + 256 * MB);     // [16,64,64] bf16    128 KB
    float* st   = (float*)(ws + 257 * MB);     // stats + fused params
    float *mY1 = st,        *rY1 = st + 256;
    float *mSc = st + 512,  *rSc = st + 768;
    float *mBr = st + 1024, *rBr = st + 1280;
    float *mF  = st + 1536, *rF  = st + 1792;
    float *gcat = st + 2048, *bcat = st + 2304;

    dim3 tb(32, 8);

    // Stage 0: x (NCHW fp32) -> xt [m,128] bf16
    transpose_cvt<<<dim3(2048, 4), tb, 0, stream>>>(x, 1, 128, 0, 128, xt,
                                                    nullptr, nullptr, nullptr, nullptr, 0);
    // Stage 1: stem GEMMs (256x128) + BN
    gemm_wmma<<<dim3(128, 8), 256, 0, stream>>>(W1,   128, xt, 128, 0, bufA, 0);
    gemm_wmma<<<dim3(128, 8), 256, 0, stream>>>(W_sc, 128, xt, 128, 0, bufR, 0);
    bn_stats<<<256, 256, 0, stream>>>(bufA, mY1, rY1);
    bn_stats<<<256, 256, 0, stream>>>(bufR, mSc, rSc);
    transpose_cvt<<<dim3(2048, 8), tb, 0, stream>>>(bufA, 0, 0, 0, 256, ht,
                                                    mY1, rY1, g1, b1, 1);
    bn_inplace<<<2048, 256, 0, stream>>>(bufR, mSc, rSc, g_sc, b_sc);

    // Branch 0: 1x1 conv (64x64) -> bufA rows 0..63
    gemm_wmma<<<dim3(128, 2), 256, 0, stream>>>(Wb0, 64, ht, 256, 0, bufA, 0);
    // Branch 1: 3x3 conv -> bufA rows 64..127
    conv3x3_wmma<<<dim3(512, 2), 256, 0, stream>>>(Wb1, ht, bufA, 64, 64);

    // Branches 2 & 3: linear global attention -> bufA rows 128..191 / 192..255
    for (int br = 0; br < 2; ++br) {
        const float* Wqkv  = br ? Wqkv3  : Wqkv2;
        const float* Wproj = br ? Wproj3 : Wproj2;
        int coff = 128 + 64 * br;
        gemm_wmma<<<dim3(128, 6), 256, 0, stream>>>(Wqkv, 64, ht, 256, coff, qkv, 0);
        transpose_cvt<<<dim3(2048, 2), tb, 0, stream>>>(qkv, 0, 0, 0, 64, qT,
                                                        nullptr, nullptr, nullptr, nullptr, 0);
        softmax_rows<<<1024, 256, 0, stream>>>(qkv, kbf);
        cvt_f32_bf16<<<2048, 256, 0, stream>>>(qkv + (size_t)128 * MTOT, vbf, (size_t)64 * MTOT);
        ctx_gemm<<<32, 256, 0, stream>>>(kbf, vbf, ctxT);
        att_gemm<<<512, 256, 0, stream>>>(qT, ctxT, attb);
        gemm_wmma<<<dim3(128, 2), 256, 0, stream>>>(Wproj, 64, attb, 64, 0, bufA, coff);
    }

    // Concat BN params, BN+ReLU all branches -> ot [m,256] bf16
    bn_stats<<<256, 256, 0, stream>>>(bufA, mBr, rBr);
    concat4<<<1, 256, 0, stream>>>(gb0, gb1, ga2, ga3, gcat);
    concat4<<<1, 256, 0, stream>>>(bb0, bb1, ba2, ba3, bcat);
    transpose_cvt<<<dim3(2048, 8), tb, 0, stream>>>(bufA, 0, 0, 0, 256, ot,
                                                    mBr, rBr, gcat, bcat, 1);

    // Final GEMM (256x256) + BN + residual + ReLU
    gemm_wmma<<<dim3(128, 8), 256, 0, stream>>>(W2, 256, ot, 256, 0, bufA, 0);
    bn_stats<<<256, 256, 0, stream>>>(bufA, mF, rF);
    final_fuse<<<4096, 256, 0, stream>>>(bufA, bufR, mF, rF, g2, b2, (float*)d_out);
}
